// FlatXGBackpropLayer_76398878261908
// MI455X (gfx1250) — matrix operand order repaired and checked by reading, not compiled, not run on hardware
//
#include <hip/hip_runtime.h>
#include <hip/hip_bf16.h>

typedef __attribute__((ext_vector_type(16))) _Float16 v16h;
typedef __attribute__((ext_vector_type(8)))  float    v8f;

#define NB      2048   // samples
#define NT      128    // trees
#define NF      256    // features
#define NNODES  127    // nodes stored per tree
#define NLVL    6      // decision levels actually used (nodes 0..62)
#define NLEAF   64     // leaves per tree

// ---------------------------------------------------------------------------
// Kernel 1: decode one-hot rows -> split feature index, via WMMA dot-with-iota.
// idx[t,n] = sum_f w[t,n,f] * f  (exact: one-hot * integer <= 255 in fp16/f32)
// One wave (one block) handles a 16-row tile; all 32 b128 loads of the tile
// slice are issued before the WMMA chain for maximal memory-level parallelism.
// Only rows 0..63 per tree are streamed (8.4 MB of the 16.6 MB tensor).
// ---------------------------------------------------------------------------
__global__ __launch_bounds__(32)
void xgb_decode_wmma(const float* __restrict__ w, int* __restrict__ idx_out)
{
    const int lane    = threadIdx.x;
    const int t       = blockIdx.x >> 2;     // tree 0..127
    const int rt      = blockIdx.x & 3;      // row-tile 0..3
    const int rowBase = rt << 4;
    const int hi      = (lane >= 16) ? 1 : 0;

    // A-matrix row for this lane (ISA 16-bit A 16x32 layout: row = lane%16;
    // lane<16 holds K0+{0..7,16..23}, lane>=16 holds K0+{8..15,24..31})
    const float* rp = w + ((size_t)t * NNODES + (rowBase + (lane & 15))) * NF;

    // Issue all 32 independent b128 loads first (LOADcnt max 63).
    float4 raw[32];
    #pragma unroll
    for (int kt = 0; kt < 8; ++kt) {
        const float4* p = (const float4*)(rp + kt * 32);
        raw[kt * 4 + 0] = p[hi ? 2 : 0];
        raw[kt * 4 + 1] = p[hi ? 3 : 1];
        raw[kt * 4 + 2] = p[hi ? 6 : 4];
        raw[kt * 4 + 3] = p[hi ? 7 : 5];
    }

    // Pre-build the 8 iota B fragments (B 32x16 fp16: lanes 0-15 hold K={0..7,16..23},
    // lanes 16-31 hold K={8..15,24..31}; every column identical = global feature id).
    v16h bv[8];
    #pragma unroll
    for (int kt = 0; kt < 8; ++kt)
        #pragma unroll
        for (int i = 0; i < 16; ++i)
            bv[kt][i] = (_Float16)(short)(kt * 32 + (hi ? 8 : 0) + (i < 8 ? i : i + 8));

    v8f c = {};
    #pragma unroll
    for (int kt = 0; kt < 8; ++kt) {
        v16h a;
        #pragma unroll
        for (int q = 0; q < 4; ++q) {
            const float4 f = raw[kt * 4 + q];
            a[q * 4 + 0] = (_Float16)f.x;
            a[q * 4 + 1] = (_Float16)f.y;
            a[q * 4 + 2] = (_Float16)f.z;
            a[q * 4 + 3] = (_Float16)f.w;
        }
        c = __builtin_amdgcn_wmma_f32_16x16x32_f16(
                false, a, false, bv[kt], (short)0, c, false, false);
    }

    // D layout: lane 0 -> M=r (r=0..7), lane 16 -> M=r+8, column N=0.
    if ((lane & 15) == 0) {
        const int mbase = rowBase + (hi ? 8 : 0);
        #pragma unroll
        for (int r = 0; r < 8; ++r)
            idx_out[t * NLEAF + mbase + r] = (int)(c[r] + 0.5f);
    }
}

// ---------------------------------------------------------------------------
// Kernel 2: oblivious-tree traversal from LDS-resident metadata.
// Block = 16 samples x 128 trees; 16 threads/sample x 8 trees each.
// LDS: 16 KB x-tile + 64 KB packed nodes + 32 KB leaves = 112 KB
// (2 workgroups per 320 KB WGP).
// ---------------------------------------------------------------------------
__global__ __launch_bounds__(256)
void xgb_forest_walk(const float* __restrict__ x,
                     const float* __restrict__ biases,
                     const float* __restrict__ leaves,
                     const float* __restrict__ base_score,
                     const int*   __restrict__ idx,
                     float* __restrict__ out)
{
    __shared__ float  xs[16 * NF];          // 16 KB: x tile
    __shared__ float2 nodes[NT * NLEAF];    // 64 KB: {bias, feat} per node
    __shared__ float  lv[NT * NLEAF];       // 32 KB: leaves

    const int tid = threadIdx.x;
    const int b0  = blockIdx.x * 16;

    // stage x tile (1024 float4) and leaves (2048 float4)
    const float4* xg = (const float4*)(x + (size_t)b0 * NF);
    float4*       xl = (float4*)xs;
    #pragma unroll
    for (int i = tid; i < 1024; i += 256) xl[i] = xg[i];
    const float4* lg = (const float4*)leaves;
    float4*       ll = (float4*)lv;
    #pragma unroll
    for (int i = tid; i < 2048; i += 256) ll[i] = lg[i];

    // stage packed node records {bias, feature} (one ds_load_b64 per level)
    for (int i = tid; i < NT * NLEAF; i += 256) {
        const int t = i >> 6, n = i & 63;
        const float bsv = (n < 63) ? biases[t * NNODES + n] : 0.0f;
        nodes[i] = make_float2(bsv, __int_as_float(idx[i]));
    }
    __syncthreads();

    const int s  = tid >> 4;     // sample within tile (0..15)
    const int tg = tid & 15;     // tree-group within sample
    const float* xrow = xs + s * NF;

    float acc = 0.0f;
    #pragma unroll
    for (int j = 0; j < 8; ++j) {            // 8 independent walk chains
        const int t = (tg << 3) + j;
        int node = 0;
        #pragma unroll
        for (int l = 0; l < NLVL; ++l) {
            const float2 nd = nodes[(t << 6) + node];
            const int f = __float_as_int(nd.y);
            node = 2 * node + 1 + ((xrow[f] > nd.x) ? 1 : 0);
        }
        acc += lv[(t << 6) + node - 63];     // node in [63,126] -> leaf
    }

    // reduce the 16-lane sample group (wave32; groups are lane-contiguous)
    acc += __shfl_xor(acc, 1, 32);
    acc += __shfl_xor(acc, 2, 32);
    acc += __shfl_xor(acc, 4, 32);
    acc += __shfl_xor(acc, 8, 32);
    if (tg == 0) out[b0 + s] = acc + base_score[0];
}

extern "C" void kernel_launch(void* const* d_in, const int* in_sizes, int n_in,
                              void* d_out, int out_size, void* d_ws, size_t ws_size,
                              hipStream_t stream)
{
    const float* x      = (const float*)d_in[0];   // (2048, 256)
    const float* w      = (const float*)d_in[1];   // (128, 127, 256) one-hot
    const float* biases = (const float*)d_in[2];   // (128, 127)
    const float* leaves = (const float*)d_in[3];   // (128, 64)
    const float* base   = (const float*)d_in[4];   // (1,)
    // d_in[5] = n_layers (constant 7, baked in)

    int* idx = (int*)d_ws;                         // 128*64*4 = 32 KB scratch

    xgb_decode_wmma<<<512, 32, 0, stream>>>(w, idx);
    xgb_forest_walk<<<128, 256, 0, stream>>>(x, biases, leaves, base, idx,
                                             (float*)d_out);
}
